// SOM_22986664968269
// MI455X (gfx1250) — matrix operand order repaired and checked by reading, not compiled, not run on hardware
//
#include <hip/hip_runtime.h>
#include <hip/hip_bf16.h>

typedef __attribute__((ext_vector_type(16))) __bf16 v16bf;
typedef __attribute__((ext_vector_type(8)))  float  v8f;

static constexpr int Bn = 65536;
static constexpr int Dn = 256;
static constexpr int Nn = 1024;

#define C_LR     0.3f
#define C_AT     0.0f
#define C_DSBETA 1e-4f
#define C_EPSDS  1e-2f

// ---------------------------------------------------------------- init
__global__ void som_init(float* counts, float* sums, int* has,
                         unsigned* mxb, unsigned* mnb, float* avgd,
                         int* nhas, float* loss) {
  int i = blockIdx.x * blockDim.x + threadIdx.x;
  int total = Nn * Dn;
  for (int idx = i; idx < total; idx += gridDim.x * blockDim.x) {
    sums[idx] = 0.f;
    if (idx < Nn) { counts[idx] = 0.f; has[idx] = 0; }
    if (idx < Dn) { mxb[idx] = 0x007FFFFFu;   // key(-inf)
                    mnb[idx] = 0xFF800000u;   // key(+inf)
                    avgd[idx] = 0.f; }
    if (idx == 0) { *nhas = 0; *loss = 0.f; }
  }
}

// ------------------------------------------------- per-node prep (rel_sum, |w|^2, bf16 split of W)
__global__ __launch_bounds__(256)
void som_prep(const float* __restrict__ weights, const float* __restrict__ relevance,
              float* __restrict__ rel_sum, float* __restrict__ wnorm,
              __bf16* __restrict__ wh, __bf16* __restrict__ wl) {
  int n = blockIdx.x, t = threadIdx.x;
  __shared__ float sr[256], sw[256];
  size_t o = (size_t)n * Dn + t;
  float w = weights[o];
  float r = relevance[o];
  sr[t] = r; sw[t] = w * w;
  __bf16 h = (__bf16)w;
  wh[o] = h;
  wl[o] = (__bf16)(w - (float)h);
  __syncthreads();
  for (int s = 128; s > 0; s >>= 1) {
    if (t < s) { sr[t] += sr[t + s]; sw[t] += sw[t + s]; }
    __syncthreads();
  }
  if (t == 0) { rel_sum[n] = sr[0]; wnorm[n] = sw[0]; }
}

// ---------------------------- GEMM (split-bf16 WMMA) + activation + argmax + fused segment sums
__global__ __launch_bounds__(128)
void som_gemm_argmax(const float* __restrict__ x,
                     const __bf16* __restrict__ wh, const __bf16* __restrict__ wl,
                     const float* __restrict__ rel_sum, const float* __restrict__ wnorm,
                     const float* __restrict__ node_control,
                     float* __restrict__ counts, float* __restrict__ sums) {
  const int b0   = blockIdx.x * 16;
  const int tid  = threadIdx.x;
  const int wave = tid >> 5;
  const int lane = tid & 31;
  const int lrow = lane & 15;   // row (A) / col (B) within 16-tile
  const int hgrp = lane >> 4;   // half-wave group

  __shared__ float s_part[128];
  __shared__ float s_xn[16];
  __shared__ float s_val[4][16];
  __shared__ int   s_idx[4][16];
  __shared__ int   s_fidx[16];
  __shared__ float s_fhigh[16];

  // ||x_b||^2 for the 16 rows of this block
  {
    int row = tid >> 3, seg = tid & 7;
    const float* xr = x + (size_t)(b0 + row) * Dn + seg * 32;
    float s = 0.f;
    #pragma unroll
    for (int j = 0; j < 32; ++j) { float v = xr[j]; s += v * v; }
    s_part[tid] = s;
  }
  __syncthreads();
  if (tid < 16) {
    float s = 0.f;
    #pragma unroll
    for (int j = 0; j < 8; ++j) s += s_part[tid * 8 + j];
    s_xn[tid] = s;
  }
  __syncthreads();

  // A fragments (ISA 16-bit A 16x32 layout): lane pair (m, m+16) shares row m.
  // elems 0..15 = K = k0 + hgrp*16 + j (contiguous, matches B)
  v16bf ah[8], al[8];
  {
    const float* xr = x + (size_t)(b0 + lrow) * Dn;
    #pragma unroll
    for (int s = 0; s < 8; ++s) {
      int k0 = s * 32 + hgrp * 16;
      #pragma unroll
      for (int j = 0; j < 8; ++j) {
        float f = xr[k0 + j];
        __bf16 h = (__bf16)f;
        ah[s][j] = h;  al[s][j] = (__bf16)(f - (float)h);
      }
      #pragma unroll
      for (int j = 0; j < 8; ++j) {
        float f = xr[k0 + 8 + j];
        __bf16 h = (__bf16)f;
        ah[s][8 + j] = h;  al[s][8 + j] = (__bf16)(f - (float)h);
      }
    }
  }

  float myxn[8];
  #pragma unroll
  for (int v = 0; v < 8; ++v) myxn[v] = s_xn[v + 8 * hgrp];

  float bestv[8];
  int   besti[8];
  #pragma unroll
  for (int v = 0; v < 8; ++v) { bestv[v] = -3.4e38f; besti[v] = 0; }

  // Each wave sweeps node tiles t = wave, wave+4, ...
  for (int t = wave; t < Nn / 16; t += 4) {
    const int col = t * 16 + lrow;                 // B column == node id
    const __bf16* whc = wh + (size_t)col * Dn;
    const __bf16* wlc = wl + (size_t)col * Dn;
    v8f acc = {};
    #pragma unroll
    for (int s = 0; s < 8; ++s) {
      // B layout: lanes 0-15 K=k0..k0+15 ; lanes 16-31 K=k0+16..k0+31 (contiguous 32B)
      v16bf bh = *(const v16bf*)(whc + s * 32 + hgrp * 16);
      v16bf bl = *(const v16bf*)(wlc + s * 32 + hgrp * 16);
      acc = __builtin_amdgcn_wmma_f32_16x16x32_bf16(false, ah[s], false, bh,
                                                    (short)0, acc, false, false);
      acc = __builtin_amdgcn_wmma_f32_16x16x32_bf16(false, ah[s], false, bl,
                                                    (short)0, acc, false, false);
      acc = __builtin_amdgcn_wmma_f32_16x16x32_bf16(false, al[s], false, bh,
                                                    (short)0, acc, false, false);
    }
    const float rs = rel_sum[col];
    const float wn = wnorm[col];
    const float nc = node_control[col];
    const float scale = rs * (1.0f / (float)Dn);
    #pragma unroll
    for (int v = 0; v < 8; ++v) {
      float dot  = acc[v];
      float dist = myxn[v] + wn - 2.0f * dot;
      float dw   = dist * scale;
      if (dw != dw) dw = 0.f;                      // isnan -> 0
      float act = rs / (rs + dw + 1e-7f) * nc;
      if (act > bestv[v]) { bestv[v] = act; besti[v] = col; }
    }
  }

  // reduce argmax across the 16 lanes of each half-wave
  #pragma unroll
  for (int v = 0; v < 8; ++v) {
    float bv = bestv[v]; int bi = besti[v];
    #pragma unroll
    for (int m = 8; m >= 1; m >>= 1) {
      float ov = __shfl_xor(bv, m, 32);
      int   oi = __shfl_xor(bi, m, 32);
      if (ov > bv || (ov == bv && oi < bi)) { bv = ov; bi = oi; }
    }
    bestv[v] = bv; besti[v] = bi;
  }
  if (lrow == 0) {
    #pragma unroll
    for (int v = 0; v < 8; ++v) {
      s_val[wave][hgrp * 8 + v] = bestv[v];
      s_idx[wave][hgrp * 8 + v] = besti[v];
    }
  }
  __syncthreads();

  // cross-wave reduce; one thread per sample row
  if (tid < 16) {
    float bv = s_val[0][tid]; int bi = s_idx[0][tid];
    #pragma unroll
    for (int w2 = 1; w2 < 4; ++w2) {
      float ov = s_val[w2][tid]; int oi = s_idx[w2][tid];
      if (ov > bv || (ov == bv && oi < bi)) { bv = ov; bi = oi; }
    }
    float high = (bv >= C_AT) ? 1.0f : 0.0f;
    s_fidx[tid] = bi; s_fhigh[tid] = high;
    if (high > 0.f) atomicAdd(&counts[bi], 1.0f);
  }
  __syncthreads();

  // fused segment sums: sums[idx[b]] += x[b]  (coalesced over d)
  for (int i = tid; i < 16 * Dn; i += 128) {
    int row = i >> 8, d = i & (Dn - 1);
    if (s_fhigh[row] > 0.f)
      atomicAdd(&sums[(size_t)s_fidx[row] * Dn + d], x[(size_t)(b0 + row) * Dn + d]);
  }
}

// -------------------------------------------- per-node update + masked feature reductions
__global__ __launch_bounds__(256)
void som_node_update(const float* __restrict__ weights, const float* __restrict__ moving_avg,
                     const float* __restrict__ counts, const float* __restrict__ sums,
                     int* __restrict__ has, unsigned* __restrict__ mxb, unsigned* __restrict__ mnb,
                     float* __restrict__ avgd, int* __restrict__ nhas,
                     float* __restrict__ out_avg, float* __restrict__ out_ma) {
  int n = blockIdx.x, d = threadIdx.x;
  size_t o = (size_t)n * Dn + d;
  float c = counts[n];
  bool  h = c > 0.f;
  float avg_s = sums[o] / fmaxf(c, 1.0f);
  out_avg[o] = avg_s;
  float w = weights[o], ma = moving_avg[o];
  const float a = C_LR * C_DSBETA;
  float ma_new = a * fabsf(avg_s - w) + (1.0f - a) * ma;
  out_ma[o] = h ? ma_new : ma;
  if (h) {
    unsigned u = __float_as_uint(ma_new);
    unsigned key = (u & 0x80000000u) ? ~u : (u | 0x80000000u);  // order-preserving
    atomicMax(&mxb[d], key);
    atomicMin(&mnb[d], key);
    atomicAdd(&avgd[d], ma_new);
    if (d == 0) { has[n] = 1; atomicAdd(nhas, 1); }
  }
}

// ------------------------------------------------------------ relevance / weights / loss
__global__ __launch_bounds__(256)
void som_finalize(const float* __restrict__ weights, const float* __restrict__ moving_avg,
                  const float* __restrict__ relevance, const float* __restrict__ out_avg,
                  const int* __restrict__ has, const unsigned* __restrict__ mxb,
                  const unsigned* __restrict__ mnb, const float* __restrict__ avgd,
                  const int* __restrict__ nhas,
                  float* __restrict__ out_w, float* __restrict__ out_rel,
                  float* __restrict__ loss) {
  int n = blockIdx.x, d = threadIdx.x;
  size_t o = (size_t)n * Dn + d;
  int h = has[n];
  float w = weights[o], ma = moving_avg[o];
  float avg_s = out_avg[o];
  const float a = C_LR * C_DSBETA;
  float ma_new = a * fabsf(avg_s - w) + (1.0f - a) * ma;
  unsigned kx = mxb[d], kn = mnb[d];
  float mx = __uint_as_float((kx & 0x80000000u) ? (kx & 0x7FFFFFFFu) : ~kx);
  float mn = __uint_as_float((kn & 0x80000000u) ? (kn & 0x7FFFFFFFu) : ~kn);
  float nh  = fmaxf((float)(*nhas), 1.0f);
  float avg = avgd[d] / nh;
  float z = (ma_new - avg) / (C_EPSDS * (mx - mn));
  float rel_new = 1.0f / (1.0f + expf(z));
  if (rel_new != rel_new) rel_new = 1.0f;          // isnan -> 1
  out_rel[o] = h ? rel_new : relevance[o];
  float delta = h ? C_LR * (avg_s - w) : 0.0f;
  out_w[o] = w + delta;

  __shared__ float sl[256];
  sl[d] = delta;
  __syncthreads();
  for (int s = 128; s > 0; s >>= 1) {
    if (d < s) sl[d] += sl[d + s];
    __syncthreads();
  }
  if (d == 0) atomicAdd(loss, sl[0] * (1.0f / (float)Bn));
}

// --------------------------------------------------------------------- launch
extern "C" void kernel_launch(void* const* d_in, const int* in_sizes, int n_in,
                              void* d_out, int out_size, void* d_ws, size_t ws_size,
                              hipStream_t stream) {
  const float* x            = (const float*)d_in[0];
  const float* weights      = (const float*)d_in[1];
  const float* moving_avg   = (const float*)d_in[2];
  const float* relevance    = (const float*)d_in[3];
  const float* node_control = (const float*)d_in[4];

  char* ws = (char*)d_ws;
  size_t off = 0;
  auto alloc = [&](size_t bytes) -> char* {
    char* p = ws + off;
    off = (off + bytes + 255) & ~(size_t)255;
    return p;
  };
  float*    rel_sum = (float*)   alloc(Nn * sizeof(float));
  float*    wnorm   = (float*)   alloc(Nn * sizeof(float));
  float*    counts  = (float*)   alloc(Nn * sizeof(float));
  int*      has     = (int*)     alloc(Nn * sizeof(int));
  unsigned* mxb     = (unsigned*)alloc(Dn * sizeof(unsigned));
  unsigned* mnb     = (unsigned*)alloc(Dn * sizeof(unsigned));
  float*    avgd    = (float*)   alloc(Dn * sizeof(float));
  int*      nhas    = (int*)     alloc(sizeof(int));
  __bf16*   wh      = (__bf16*)  alloc((size_t)Nn * Dn * sizeof(__bf16));
  __bf16*   wl      = (__bf16*)  alloc((size_t)Nn * Dn * sizeof(__bf16));
  float*    sums    = (float*)   alloc((size_t)Nn * Dn * sizeof(float));

  float* out      = (float*)d_out;
  float* out_avg  = out;
  float* out_w    = out + 1 * (size_t)Nn * Dn;
  float* out_ma   = out + 2 * (size_t)Nn * Dn;
  float* out_rel  = out + 3 * (size_t)Nn * Dn;
  float* out_loss = out + 4 * (size_t)Nn * Dn;

  som_init<<<1024, 256, 0, stream>>>(counts, sums, has, mxb, mnb, avgd, nhas, out_loss);
  som_prep<<<Nn, 256, 0, stream>>>(weights, relevance, rel_sum, wnorm, wh, wl);
  som_gemm_argmax<<<Bn / 16, 128, 0, stream>>>(x, wh, wl, rel_sum, wnorm, node_control,
                                               counts, sums);
  som_node_update<<<Nn, 256, 0, stream>>>(weights, moving_avg, counts, sums, has,
                                          mxb, mnb, avgd, nhas, out_avg, out_ma);
  som_finalize<<<Nn, 256, 0, stream>>>(weights, moving_avg, relevance, out_avg, has,
                                       mxb, mnb, avgd, nhas, out_w, out_rel, out_loss);
}
